// WindowMSA_14482629722426
// MI455X (gfx1250) — matrix operand order repaired and checked by reading, not compiled, not run on hardware
//
#include <hip/hip_runtime.h>

// ---------------------------------------------------------------------------
// MI455X (gfx1250) fused window-MSA pipeline.
// bf16 WMMA (v_wmma_f32_16x16x32_bf16) everywhere; f32 accumulation.
// Native __bf16 converts in epilogues (v_cvt_*_bf16_f32) instead of bit-twiddled RNE.
// ---------------------------------------------------------------------------

typedef __attribute__((ext_vector_type(16))) __bf16 v16bf;
typedef __attribute__((ext_vector_type(8)))  float  v8f;

#define TDIM 4
#define HDIM 128
#define WDIM 192
#define CDIM 256
#define HWN  (HDIM * WDIM)     // 24576 spatial positions
#define NHEAD 8
#define KD    32
#define LDP   272              // bf16 row stride for 256-wide tiles (pad vs banks)
#define SPP   72               // stride for 64-wide score/prob tiles

// LDS carve (bytes):
//   sXN/sO : 64*272*2   = 34816   [0      , 34816)
//   sQ     : 64*272*2   = 34816   [34816  , 69632)
//   sK     : 64*272*2   = 34816   [69632  , 104448)
//   sVT    : 256*72*2   = 36864   [104448 , 141312)
//   sS     : 64*72*4    = 18432   [141312 , 159744)
//   sP     : 64*72*2    =  9216   [159744 , 168960)
//   sRed   : 512*4      =  2048   [168960 , 171008)
static constexpr unsigned SMEM_BYTES = 171008;

__device__ __forceinline__ __bf16 f2bf(float f) { return (__bf16)f; }

// A fragment: 16x32 bf16, row-major source with stride ld (in elements).
// Lane l: M = m0 + (l%16); elems 0..15 -> K = k0 + (l/16)*16 + e (contiguous).
__device__ __forceinline__ v16bf load_a16x32(const __bf16* base, int ld,
                                             int m0, int k0, int lsub, int lhi) {
  const __bf16* ap = base + (m0 + lsub) * ld + k0 + lhi * 16;
  v16bf a;
#pragma unroll
  for (int e = 0; e < 16; ++e) a[e] = ap[e];
  return a;
}

// B fragment: 32x16 bf16 from an [N][K] row-major buffer with stride ld.
// Lane l: N = n0 + (l%16); elems 0..15 -> K = k0 + (l/16)*16 + e (contiguous 32B).
__device__ __forceinline__ v16bf load_b32x16(const __bf16* base, int ld,
                                             int n0, int k0, int lsub, int lhi) {
  const __bf16* bp = base + (n0 + lsub) * ld + k0 + lhi * 16;
  v16bf b;
#pragma unroll
  for (int e = 0; e < 16; ++e) b[e] = bp[e];
  return b;
}

// C = A[64x256] * W^T[256x256] ; 64 output tiles, 8 per wave; K = 8 steps of 32.
template <class Epi>
__device__ __forceinline__ void gemm_64x256(const __bf16* A, int lda,
                                            const __bf16* BT,
                                            int wave, int lsub, int lhi, Epi&& epi) {
  for (int j = 0; j < 8; ++j) {
    const int tile = wave + 8 * j;
    const int m0 = (tile >> 4) * 16;
    const int n0 = (tile & 15) * 16;
    v8f acc = {};
#pragma unroll
    for (int kt = 0; kt < 8; ++kt) {
      v16bf a = load_a16x32(A, lda, m0, kt * 32, lsub, lhi);
      v16bf b = load_b32x16(BT, 256, n0, kt * 32, lsub, lhi);
      acc = __builtin_amdgcn_wmma_f32_16x16x32_bf16(false, a, false, b,
                                                    (short)0, acc, false, false);
    }
    epi(m0, n0, acc);
  }
}

// ---------------------------------------------------------------------------
// Weight prep: f32 [K][N] -> bf16 transposed [N][K] (B-operand friendly).
// ---------------------------------------------------------------------------
struct WPtrs { const float* p[10]; };

__global__ __launch_bounds__(256) void prep_weights(WPtrs wp, __bf16* out) {
  const int g   = blockIdx.x * 256 + threadIdx.x;   // [0, 10*65536)
  const int mat = g >> 16;
  const int idx = g & 65535;
  const int n = idx >> 8, k = idx & 255;
  out[mat * 65536 + n * 256 + k] = f2bf(wp.p[mat][k * 256 + n]);
}

// ---------------------------------------------------------------------------
// Kernel A: LN1 + spatial window MSA. One block = one (t, gh, gw) window.
// ---------------------------------------------------------------------------
__global__ __launch_bounds__(256) void spatial_msa_kernel(
    const float* __restrict__ x, const float* __restrict__ ln_g,
    const float* __restrict__ ln_b,
    const __bf16* __restrict__ wqT, const __bf16* __restrict__ wkT,
    const __bf16* __restrict__ wvT, const __bf16* __restrict__ woT,
    const float* __restrict__ bq, const float* __restrict__ bk,
    const float* __restrict__ bv, const float* __restrict__ bo,
    __bf16* __restrict__ y1) {
  extern __shared__ char smem[];
  __bf16* sXN = (__bf16*)smem;
  __bf16* sQ  = sXN + 64 * LDP;
  __bf16* sK  = sQ  + 64 * LDP;
  __bf16* sVT = sK  + 64 * LDP;                  // [channel][token], stride SPP
  float*  sS  = (float*)(sVT + 256 * SPP);
  __bf16* sP  = (__bf16*)(sS + 64 * SPP);
  float*  sRed = (float*)(sP + 64 * SPP);
  __bf16* sO  = sXN;                             // alias: X dead after QKV

  const int win = blockIdx.x;
  const int t   = win / (16 * 24);
  const int rem = win % (16 * 24);
  const int gh  = rem / 24, gw = rem % 24;
  const int tid  = threadIdx.x;
  const int wave = tid >> 5, lane = tid & 31, lsub = lane & 15, lhi = lane >> 4;
  const int row = tid >> 2, qq = tid & 3;        // 4 threads per token row

  // ---- Stage 1: LayerNorm(x) -> bf16 X in LDS -------------------------------
  {
    const int mh = row >> 3, mw = row & 7;
    const int hh = gh * 8 + mh, ww = gw * 8 + mw;
    const float* xr = x + ((size_t)((t * HDIM + hh) * WDIM + ww)) * CDIM;
    float sum = 0.f, sq = 0.f;
#pragma unroll 8
    for (int c = qq * 64; c < qq * 64 + 64; ++c) {
      float v = xr[c]; sum += v; sq += v * v;
    }
    sRed[row * 4 + qq] = sum;
    sRed[256 + row * 4 + qq] = sq;
    __syncthreads();
    const float s0 = sRed[row * 4] + sRed[row * 4 + 1] + sRed[row * 4 + 2] + sRed[row * 4 + 3];
    const float s1 = sRed[256 + row * 4] + sRed[256 + row * 4 + 1] +
                     sRed[256 + row * 4 + 2] + sRed[256 + row * 4 + 3];
    const float mu   = s0 * (1.f / 256.f);
    const float rstd = rsqrtf(s1 * (1.f / 256.f) - mu * mu + 1e-3f);
#pragma unroll 8
    for (int c = qq * 64; c < qq * 64 + 64; ++c)
      sXN[row * LDP + c] = f2bf((xr[c] - mu) * rstd * ln_g[c] + ln_b[c]);
  }
  __syncthreads();

  // ---- Stage 2: Q, K, V projections (V stored transposed) -------------------
  gemm_64x256(sXN, LDP, wqT, wave, lsub, lhi, [&](int m0, int n0, v8f acc) {
    const float bb = bq[n0 + lsub];
#pragma unroll
    for (int r = 0; r < 8; ++r)
      sQ[(m0 + r + lhi * 8) * LDP + n0 + lsub] = f2bf(acc[r] + bb);
  });
  gemm_64x256(sXN, LDP, wkT, wave, lsub, lhi, [&](int m0, int n0, v8f acc) {
    const float bb = bk[n0 + lsub];
#pragma unroll
    for (int r = 0; r < 8; ++r)
      sK[(m0 + r + lhi * 8) * LDP + n0 + lsub] = f2bf(acc[r] + bb);
  });
  gemm_64x256(sXN, LDP, wvT, wave, lsub, lhi, [&](int m0, int n0, v8f acc) {
    const float bb = bv[n0 + lsub];
#pragma unroll
    for (int r = 0; r < 8; ++r)   // transposed: [channel][token]
      sVT[(n0 + lsub) * SPP + m0 + r + lhi * 8] = f2bf(acc[r] + bb);
  });
  __syncthreads();

  // ---- Stage 3: per-head attention ------------------------------------------
  const float scale = 0.17677669529663688f;      // 1/sqrt(32)
  for (int h = 0; h < NHEAD; ++h) {
    // scores: S[64x64] = Q_h[64x32] * K_h^T ; one 16x16x32 WMMA per tile
#pragma unroll
    for (int j = 0; j < 2; ++j) {
      const int tile = wave * 2 + j;
      const int m0 = (tile >> 2) * 16, t0 = (tile & 3) * 16;
      v16bf a = load_a16x32(sQ, LDP, m0, h * KD, lsub, lhi);
      v16bf b = load_b32x16(sK, LDP, t0, h * KD, lsub, lhi);
      v8f acc = {};
      acc = __builtin_amdgcn_wmma_f32_16x16x32_bf16(false, a, false, b,
                                                    (short)0, acc, false, false);
#pragma unroll
      for (int r = 0; r < 8; ++r)
        sS[(m0 + r + lhi * 8) * SPP + t0 + lsub] = acc[r];
    }
    __syncthreads();
    // softmax over 64 columns (4 threads per row)
    {
      float mx = -1e30f;
#pragma unroll 4
      for (int c = qq * 16; c < qq * 16 + 16; ++c)
        mx = fmaxf(mx, sS[row * SPP + c] * scale);
      sRed[row * 4 + qq] = mx;
      __syncthreads();
      mx = fmaxf(fmaxf(sRed[row * 4], sRed[row * 4 + 1]),
                 fmaxf(sRed[row * 4 + 2], sRed[row * 4 + 3]));
      float sum = 0.f;
#pragma unroll 4
      for (int c = qq * 16; c < qq * 16 + 16; ++c) {
        float e = __expf(sS[row * SPP + c] * scale - mx);
        sS[row * SPP + c] = e; sum += e;
      }
      sRed[256 + row * 4 + qq] = sum;
      __syncthreads();
      sum = sRed[256 + row * 4] + sRed[256 + row * 4 + 1] +
            sRed[256 + row * 4 + 2] + sRed[256 + row * 4 + 3];
      const float inv = 1.f / sum;
#pragma unroll 4
      for (int c = qq * 16; c < qq * 16 + 16; ++c)
        sP[row * SPP + c] = f2bf(sS[row * SPP + c] * inv);
    }
    __syncthreads();
    // AV: O_h[64x32] = P[64x64] * V_h[64x32] ; one tile per wave, K = 2x32
    {
      const int m0 = (wave >> 1) * 16, n0 = (wave & 1) * 16;
      v8f acc = {};
#pragma unroll
      for (int kt = 0; kt < 2; ++kt) {
        v16bf a = load_a16x32(sP, SPP, m0, kt * 32, lsub, lhi);
        v16bf b = load_b32x16(sVT, SPP, h * KD + n0, kt * 32, lsub, lhi);
        acc = __builtin_amdgcn_wmma_f32_16x16x32_bf16(false, a, false, b,
                                                      (short)0, acc, false, false);
      }
#pragma unroll
      for (int r = 0; r < 8; ++r)
        sO[(m0 + r + lhi * 8) * LDP + h * KD + n0 + lsub] = f2bf(acc[r]);
    }
    __syncthreads();
  }

  // ---- Stage 4: output projection -> y1 (bf16, [t][h][w][c]) ----------------
  gemm_64x256(sO, LDP, woT, wave, lsub, lhi, [&](int m0, int n0, v8f acc) {
    const float bb = bo[n0 + lsub];
#pragma unroll
    for (int r = 0; r < 8; ++r) {
      const int s  = m0 + r + lhi * 8;
      const int hh = gh * 8 + (s >> 3), ww = gw * 8 + (s & 7);
      const size_t tok = (size_t)t * HWN + (size_t)hh * WDIM + ww;
      y1[tok * CDIM + n0 + lsub] = f2bf(acc[r] + bb);
    }
  });
}

// ---------------------------------------------------------------------------
// Kernel B: temporal MSA (16 positions x T=4, block-diag mask) + LN2 + MLP.
// ---------------------------------------------------------------------------
__global__ __launch_bounds__(256) void temporal_mlp_kernel(
    const __bf16* __restrict__ y1,
    const __bf16* __restrict__ wqT, const __bf16* __restrict__ wkT,
    const __bf16* __restrict__ wvT, const __bf16* __restrict__ woT,
    const __bf16* __restrict__ d1T, const __bf16* __restrict__ d2T,
    const float* __restrict__ bq, const float* __restrict__ bk,
    const float* __restrict__ bv, const float* __restrict__ bo,
    const float* __restrict__ ln2g, const float* __restrict__ ln2b,
    const float* __restrict__ d1b, const float* __restrict__ d2b,
    float* __restrict__ out) {
  extern __shared__ char smem[];
  __bf16* sXN = (__bf16*)smem;
  __bf16* sQ  = sXN + 64 * LDP;
  __bf16* sK  = sQ  + 64 * LDP;
  __bf16* sVT = sK  + 64 * LDP;
  float*  sS  = (float*)(sVT + 256 * SPP);
  __bf16* sP  = (__bf16*)(sS + 64 * SPP);
  float*  sRed = (float*)(sP + 64 * SPP);
  __bf16* sO  = sXN;                 // aliases
  float*  sY2 = (float*)sQ;          // 64 x 264 f32 over sQ+sK (67584 B)
  __bf16* sZ  = sVT;                 // LN2 output (bf16) over sVT
  __bf16* sH1 = sXN;                 // MLP hidden over sXN/sO

  const int posBase = blockIdx.x * 16;
  const int tid  = threadIdx.x;
  const int wave = tid >> 5, lane = tid & 31, lsub = lane & 15, lhi = lane >> 4;
  const int row = tid >> 2, qq = tid & 3;    // row = p*4 + t

  // ---- Stage 1: gather 64 rows of y1 (bf16 copy, 128B per thread) -----------
  {
    const int p = row >> 2, tt = row & 3;
    const uint4* src = (const uint4*)(y1 + ((size_t)(tt * HWN + posBase + p)) * CDIM + qq * 64);
    uint4* dst = (uint4*)(sXN + row * LDP + qq * 64);
#pragma unroll
    for (int v = 0; v < 8; ++v) dst[v] = src[v];
  }
  __syncthreads();

  // ---- Stage 2: temporal Q, K, V --------------------------------------------
  gemm_64x256(sXN, LDP, wqT, wave, lsub, lhi, [&](int m0, int n0, v8f acc) {
    const float bb = bq[n0 + lsub];
#pragma unroll
    for (int r = 0; r < 8; ++r)
      sQ[(m0 + r + lhi * 8) * LDP + n0 + lsub] = f2bf(acc[r] + bb);
  });
  gemm_64x256(sXN, LDP, wkT, wave, lsub, lhi, [&](int m0, int n0, v8f acc) {
    const float bb = bk[n0 + lsub];
#pragma unroll
    for (int r = 0; r < 8; ++r)
      sK[(m0 + r + lhi * 8) * LDP + n0 + lsub] = f2bf(acc[r] + bb);
  });
  gemm_64x256(sXN, LDP, wvT, wave, lsub, lhi, [&](int m0, int n0, v8f acc) {
    const float bb = bv[n0 + lsub];
#pragma unroll
    for (int r = 0; r < 8; ++r)
      sVT[(n0 + lsub) * SPP + m0 + r + lhi * 8] = f2bf(acc[r] + bb);
  });
  __syncthreads();

  // ---- Stage 3: masked attention (valid iff same position block of 4) -------
  const float scale = 0.17677669529663688f;
  for (int h = 0; h < NHEAD; ++h) {
#pragma unroll
    for (int j = 0; j < 2; ++j) {
      const int tile = wave * 2 + j;
      const int m0 = (tile >> 2) * 16, t0 = (tile & 3) * 16;
      v16bf a = load_a16x32(sQ, LDP, m0, h * KD, lsub, lhi);
      v16bf b = load_b32x16(sK, LDP, t0, h * KD, lsub, lhi);
      v8f acc = {};
      acc = __builtin_amdgcn_wmma_f32_16x16x32_bf16(false, a, false, b,
                                                    (short)0, acc, false, false);
#pragma unroll
      for (int r = 0; r < 8; ++r)
        sS[(m0 + r + lhi * 8) * SPP + t0 + lsub] = acc[r];
    }
    __syncthreads();
    {
      const int blk = row >> 2;              // my position block
      float mx = -1e30f;
#pragma unroll 4
      for (int c = qq * 16; c < qq * 16 + 16; ++c)
        if ((c >> 2) == blk) mx = fmaxf(mx, sS[row * SPP + c] * scale);
      sRed[row * 4 + qq] = mx;
      __syncthreads();
      mx = fmaxf(fmaxf(sRed[row * 4], sRed[row * 4 + 1]),
                 fmaxf(sRed[row * 4 + 2], sRed[row * 4 + 3]));
      float sum = 0.f;
#pragma unroll 4
      for (int c = qq * 16; c < qq * 16 + 16; ++c) {
        float e = ((c >> 2) == blk) ? __expf(sS[row * SPP + c] * scale - mx) : 0.f;
        sS[row * SPP + c] = e; sum += e;
      }
      sRed[256 + row * 4 + qq] = sum;
      __syncthreads();
      sum = sRed[256 + row * 4] + sRed[256 + row * 4 + 1] +
            sRed[256 + row * 4 + 2] + sRed[256 + row * 4 + 3];
      const float inv = 1.f / sum;
#pragma unroll 4
      for (int c = qq * 16; c < qq * 16 + 16; ++c)
        sP[row * SPP + c] = f2bf(sS[row * SPP + c] * inv);
    }
    __syncthreads();
    {
      const int m0 = (wave >> 1) * 16, n0 = (wave & 1) * 16;
      v8f acc = {};
#pragma unroll
      for (int kt = 0; kt < 2; ++kt) {
        v16bf a = load_a16x32(sP, SPP, m0, kt * 32, lsub, lhi);
        v16bf b = load_b32x16(sVT, SPP, h * KD + n0, kt * 32, lsub, lhi);
        acc = __builtin_amdgcn_wmma_f32_16x16x32_bf16(false, a, false, b,
                                                      (short)0, acc, false, false);
      }
#pragma unroll
      for (int r = 0; r < 8; ++r)
        sO[(m0 + r + lhi * 8) * LDP + h * KD + n0 + lsub] = f2bf(acc[r]);
    }
    __syncthreads();
  }

  // ---- Stage 4: output projection -> sY2 (f32) ------------------------------
  gemm_64x256(sO, LDP, woT, wave, lsub, lhi, [&](int m0, int n0, v8f acc) {
    const float bb = bo[n0 + lsub];
#pragma unroll
    for (int r = 0; r < 8; ++r)
      sY2[(m0 + r + lhi * 8) * 264 + n0 + lsub] = acc[r] + bb;
  });
  __syncthreads();

  // ---- Stage 5: LN2 -> bf16 Z -----------------------------------------------
  {
    float sum = 0.f, sq = 0.f;
#pragma unroll 8
    for (int c = qq * 64; c < qq * 64 + 64; ++c) {
      float v = sY2[row * 264 + c]; sum += v; sq += v * v;
    }
    sRed[row * 4 + qq] = sum;
    sRed[256 + row * 4 + qq] = sq;
    __syncthreads();
    const float s0 = sRed[row * 4] + sRed[row * 4 + 1] + sRed[row * 4 + 2] + sRed[row * 4 + 3];
    const float s1 = sRed[256 + row * 4] + sRed[256 + row * 4 + 1] +
                     sRed[256 + row * 4 + 2] + sRed[256 + row * 4 + 3];
    const float mu   = s0 * (1.f / 256.f);
    const float rstd = rsqrtf(s1 * (1.f / 256.f) - mu * mu + 1e-3f);
#pragma unroll 8
    for (int c = qq * 64; c < qq * 64 + 64; ++c)
      sZ[row * LDP + c] = f2bf((sY2[row * 264 + c] - mu) * rstd * ln2g[c] + ln2b[c]);
  }
  __syncthreads();

  // ---- Stage 6: H1 = relu(Z @ d1 + b1) --------------------------------------
  gemm_64x256(sZ, LDP, d1T, wave, lsub, lhi, [&](int m0, int n0, v8f acc) {
    const float bb = d1b[n0 + lsub];
#pragma unroll
    for (int r = 0; r < 8; ++r)
      sH1[(m0 + r + lhi * 8) * LDP + n0 + lsub] = f2bf(fmaxf(acc[r] + bb, 0.f));
  });
  __syncthreads();

  // ---- Stage 7: out = relu(H1 @ d2 + b2), scatter f32 -----------------------
  gemm_64x256(sH1, LDP, d2T, wave, lsub, lhi, [&](int m0, int n0, v8f acc) {
    const float bb = d2b[n0 + lsub];
#pragma unroll
    for (int r = 0; r < 8; ++r) {
      const int s = m0 + r + lhi * 8;
      const int p = s >> 2, tt = s & 3;
      out[((size_t)(tt * HWN + posBase + p)) * CDIM + n0 + lsub] =
          fmaxf(acc[r] + bb, 0.f);
    }
  });
}

// ---------------------------------------------------------------------------
extern "C" void kernel_launch(void* const* d_in, const int* in_sizes, int n_in,
                              void* d_out, int out_size, void* d_ws, size_t ws_size,
                              hipStream_t stream) {
  const float* x     = (const float*)d_in[0];
  const float* ln1_g = (const float*)d_in[1];
  const float* ln1_b = (const float*)d_in[2];
  const float* sp_bq = (const float*)d_in[4];
  const float* sp_bk = (const float*)d_in[6];
  const float* sp_bv = (const float*)d_in[8];
  const float* sp_bo = (const float*)d_in[10];
  const float* tp_bq = (const float*)d_in[12];
  const float* tp_bk = (const float*)d_in[14];
  const float* tp_bv = (const float*)d_in[16];
  const float* tp_bo = (const float*)d_in[18];
  const float* ln2_g = (const float*)d_in[19];
  const float* ln2_b = (const float*)d_in[20];
  const float* d1_b  = (const float*)d_in[22];
  const float* d2_b  = (const float*)d_in[24];

  __bf16* wT = (__bf16*)d_ws;                      // 10 x 256x256 bf16 (transposed)
  __bf16* y1 = wT + 10 * 65536;                    // 98304 x 256 bf16

  WPtrs wp;
  wp.p[0] = (const float*)d_in[3];   // sp_wq
  wp.p[1] = (const float*)d_in[5];   // sp_wk
  wp.p[2] = (const float*)d_in[7];   // sp_wv
  wp.p[3] = (const float*)d_in[9];   // sp_wo
  wp.p[4] = (const float*)d_in[11];  // tp_wq
  wp.p[5] = (const float*)d_in[13];  // tp_wk
  wp.p[6] = (const float*)d_in[15];  // tp_wv
  wp.p[7] = (const float*)d_in[17];  // tp_wo
  wp.p[8] = (const float*)d_in[21];  // d1_w
  wp.p[9] = (const float*)d_in[23];  // d2_w

  prep_weights<<<2560, 256, 0, stream>>>(wp, wT);
  spatial_msa_kernel<<<1536, 256, SMEM_BYTES, stream>>>(
      x, ln1_g, ln1_b,
      wT, wT + 65536, wT + 2 * 65536, wT + 3 * 65536,
      sp_bq, sp_bk, sp_bv, sp_bo, y1);
  temporal_mlp_kernel<<<1536, 256, SMEM_BYTES, stream>>>(
      y1,
      wT + 4 * 65536, wT + 5 * 65536, wT + 6 * 65536, wT + 7 * 65536,
      wT + 8 * 65536, wT + 9 * 65536,
      tp_bq, tp_bk, tp_bv, tp_bo, ln2_g, ln2_b, d1_b, d2_b,
      (float*)d_out);
}
